// MultiHeadAttention_71090298683569
// MI455X (gfx1250) — matrix operand order repaired and checked by reading, not compiled, not run on hardware
//
#include <hip/hip_runtime.h>
#include <math.h>

// Problem constants (reference: B=2, S=2048, D=1024, H=16, DK=64)
#define B_  2
#define S_  2048
#define D_  1024
#define H_  16
#define DK_ 64

typedef __attribute__((ext_vector_type(16))) __bf16 v16bf;
typedef __attribute__((ext_vector_type(8)))  __bf16 v8bf;
typedef __attribute__((ext_vector_type(8)))  float  v8f;
typedef __attribute__((ext_vector_type(4)))  int    v4i;

// ---------------------------------------------------------------------------
// CDNA5 async global->LDS staging (GLOBAL_LOAD_ASYNC_TO_LDS_B128, ASYNCcnt).
// Builtin signature (from compiler diagnostic): (v4i AS1* src, v4i AS3* dst,
// imm offset, imm cpol). Guarded; falls back to a synchronous staging copy.
// ---------------------------------------------------------------------------
#if defined(__gfx1250__) && \
    __has_builtin(__builtin_amdgcn_global_load_async_to_lds_b128) && \
    __has_builtin(__builtin_amdgcn_s_wait_asynccnt)
#define CDNA5_ASYNC 1
#else
#define CDNA5_ASYNC 0
#endif

__device__ __forceinline__ void async_copy_b128(void* lds_dst, const void* gsrc) {
#if CDNA5_ASYNC
  typedef __attribute__((address_space(1))) v4i as1_v4i;
  typedef __attribute__((address_space(3))) v4i as3_v4i;
  __builtin_amdgcn_global_load_async_to_lds_b128(
      (as1_v4i*)(v4i*)(void*)gsrc,
      (as3_v4i*)(v4i*)lds_dst, 0, 0);
#else
  *(uint4*)lds_dst = *(const uint4*)gsrc;
#endif
}

template <int N>
__device__ __forceinline__ void wait_async() {
#if CDNA5_ASYNC
  __builtin_amdgcn_s_wait_asynccnt(N);
#endif
}

// CDNA5 WMMA f32 = bf16 A(16x32) * bf16 B(32x16) + f32 C(16x16)
__device__ __forceinline__ v8f wmma_bf16(v16bf a, v16bf b, v8f c) {
  return __builtin_amdgcn_wmma_f32_16x16x32_bf16(
      /*neg_a=*/false, a, /*neg_b=*/false, b,
      /*c_mod=*/(short)0, c, /*reuse_a=*/false, /*reuse_b=*/false);
}

// A-matrix (16x32, 16-bit) intra-lane K index per ISA 7.12.2:
// lanes 0-15 hold K={0..7,16..23}, lanes 16-31 hold K={8..15,24..31}
__device__ __forceinline__ int a_kmap(int e, int half) {
  return (e < 8) ? (e + 8 * half) : (e + 8 + 8 * half);
}

// ---------------------------------------------------------------------------
// Kernel 1: P = X @ W^T + b (M=4096,N=1024,K=1024), f32 in, bf16 head-split out.
// Block = 8 waves x (16 rows x 64 cols). The 64x32 W tile is shared by all 8
// waves -> stage it in LDS via async b128 copies, double-buffered.
// ---------------------------------------------------------------------------
__global__ __launch_bounds__(256)
void proj_bf16_kernel(const float* __restrict__ X, const float* __restrict__ W,
                      const float* __restrict__ bias, __bf16* __restrict__ dst) {
  __shared__ float lw[2][64][32];  // [buf][n_local][k_local], 16 KB

  const int tid  = threadIdx.x;
  const int lane = tid & 31;
  const int wave = tid >> 5;
  const int ng   = blockIdx.x & 15;  // N-group of 64
  const int mblk = blockIdx.x >> 4;  // 8 M-tiles per block
  const int m0   = (mblk * 8 + wave) * 16;
  const int n0   = ng * 64;
  const int half = lane >> 4, mr = lane & 15;

  // Stage W[n0+row][kk..kk+31] -> lw[buf]; 512 b128 chunks / 256 threads = 2.
  auto stageW = [&](int kk, int buf) {
#pragma unroll
    for (int j = 0; j < 2; ++j) {
      int cid = tid + 256 * j;      // 0..511
      int row = cid >> 3, c = cid & 7;
      async_copy_b128(&lw[buf][row][c * 4],
                      &W[(size_t)(n0 + row) * D_ + kk + c * 4]);
    }
  };

  v8f acc[4] = {};
  stageW(0, 0);

  for (int kk = 0; kk < D_; kk += 32) {
    const int buf = (kk >> 5) & 1;
    if (kk + 32 < D_) { stageW(kk + 32, buf ^ 1); wait_async<2>(); }
    else              { wait_async<0>(); }
    __syncthreads();

    // A fragment: the a_kmap layout is two contiguous 8-float runs per lane.
    const float* rowp = &X[(size_t)(m0 + mr) * D_ + kk];
    float4 a0 = *(const float4*)(rowp + 8 * half);
    float4 a1 = *(const float4*)(rowp + 8 * half + 4);
    float4 a2 = *(const float4*)(rowp + 16 + 8 * half);
    float4 a3 = *(const float4*)(rowp + 20 + 8 * half);
    v16bf a;
    a[0]=(__bf16)a0.x;  a[1]=(__bf16)a0.y;  a[2]=(__bf16)a0.z;  a[3]=(__bf16)a0.w;
    a[4]=(__bf16)a1.x;  a[5]=(__bf16)a1.y;  a[6]=(__bf16)a1.z;  a[7]=(__bf16)a1.w;
    a[8]=(__bf16)a2.x;  a[9]=(__bf16)a2.y;  a[10]=(__bf16)a2.z; a[11]=(__bf16)a2.w;
    a[12]=(__bf16)a3.x; a[13]=(__bf16)a3.y; a[14]=(__bf16)a3.z; a[15]=(__bf16)a3.w;

#pragma unroll
    for (int t = 0; t < 4; ++t) {
      // B[k][n] = W[n][k]: n = lane&15, e -> k via a_kmap; from LDS.
      v16bf bfr;
#pragma unroll
      for (int e = 0; e < 16; ++e)
        bfr[e] = (__bf16)lw[buf][t * 16 + mr][a_kmap(e, half)];
      acc[t] = wmma_bf16(a, bfr, acc[t]);
    }
    __syncthreads();
  }

  // C layout: vgpr r, lane -> m = r + 8*(lane>>4), n = lane&15
#pragma unroll
  for (int t = 0; t < 4; ++t) {
    int n = n0 + t * 16 + (lane & 15);
    int h = n >> 6, dk = n & 63;
    float bn = bias[n];
#pragma unroll
    for (int r = 0; r < 8; ++r) {
      int m = m0 + r + 8 * half;
      int b = m >> 11, s = m & (S_ - 1);
      dst[((size_t)(b * H_ + h) * S_ + s) * DK_ + dk] = (__bf16)(acc[t][r] + bn);
    }
  }
}

// ---------------------------------------------------------------------------
// Kernel 2: fused per-head  scores=(Q K^T)/8 -> mask -> qkv = scores @ V
//           -> softmax over DK -> X (bf16). Scores never touch HBM.
// Block = 4 waves sharing one (b,h); the 32-row K/V band is shared by all
// 4 waves -> async-stage K and V bands into LDS, double-buffered.
// ---------------------------------------------------------------------------
__global__ __launch_bounds__(128)
void attn_fused_kernel(const __bf16* __restrict__ Q, const __bf16* __restrict__ K,
                       const __bf16* __restrict__ V, const int* __restrict__ mask,
                       __bf16* __restrict__ X) {
  __shared__ __bf16 lk[2][32][DK_];  // 8 KB
  __shared__ __bf16 lv[2][32][DK_];  // 8 KB
  __shared__ float  sc[4][16][33];   // per-wave score reshape scratch

  const int tid  = threadIdx.x;
  const int lane = tid & 31;
  const int wave = tid >> 5;
  const int bh   = blockIdx.x >> 5;  // b*H + h
  const int qblk = blockIdx.x & 31;
  const int b    = bh >> 4;
  const int h    = bh & 15;
  const int q0   = (qblk * 4 + wave) * 16;
  const int half = lane >> 4, mr = lane & 15;

  const __bf16* Qb = Q + (size_t)bh * S_ * DK_;
  const __bf16* Kb = K + (size_t)bh * S_ * DK_;
  const __bf16* Vb = V + (size_t)bh * S_ * DK_;

  // Stage K and V rows kk..kk+31 (4 KB each) -> 512 b128 chunks / 128 thr = 4.
  auto stageKV = [&](int kk, int buf) {
#pragma unroll
    for (int j = 0; j < 4; ++j) {
      int cid = tid + 128 * j;       // 0..511
      int mat = cid >> 8;            // 0 = K, 1 = V
      int q   = cid & 255;
      int row = q >> 3, c = q & 7;
      const __bf16* g = mat ? &Vb[(size_t)(kk + row) * DK_ + c * 8]
                            : &Kb[(size_t)(kk + row) * DK_ + c * 8];
      __bf16* l = mat ? &lv[buf][row][c * 8] : &lk[buf][row][c * 8];
      async_copy_b128(l, g);
    }
  };

  // Loop-invariant A fragments of the Q tile (vectorized: 2x 16B per band).
  const __bf16* qrow = &Qb[(size_t)(q0 + mr) * DK_];
  v16bf aq[2];
#pragma unroll
  for (int band = 0; band < 2; ++band) {
    v8bf c0 = *(const v8bf*)(qrow + band * 32 + 8 * half);
    v8bf c1 = *(const v8bf*)(qrow + band * 32 + 16 + 8 * half);
#pragma unroll
    for (int e = 0; e < 8; ++e) { aq[band][e] = c0[e]; aq[band][e + 8] = c1[e]; }
  }

  v8f acc[4] = {};
  stageKV(0, 0);

  for (int kk = 0; kk < S_; kk += 32) {
    const int buf = (kk >> 5) & 1;
    if (kk + 32 < S_) { stageKV(kk + 32, buf ^ 1); wait_async<4>(); }
    else              { wait_async<0>(); }
    __syncthreads();

    // Prefetch the band after next into GL2 (global_prefetch_b8).
    if (kk + 64 < S_) {
      __builtin_prefetch(&Kb[(size_t)(kk + 64 + lane) * DK_], 0, 1);
      __builtin_prefetch(&Vb[(size_t)(kk + 64 + lane) * DK_], 0, 1);
    }

    // ---- scores for the two 16-wide key tiles of this band ----
#pragma unroll
    for (int tile = 0; tile < 2; ++tile) {
      v8f s = {};
#pragma unroll
      for (int band = 0; band < 2; ++band) {
        // B[k=d][n=key row]: n = lane&15 (key row in tile), e -> d via a_kmap.
        v16bf bk;
#pragma unroll
        for (int e = 0; e < 16; ++e)
          bk[e] = lk[buf][tile * 16 + mr][band * 32 + a_kmap(e, half)];
        s = wmma_bf16(aq[band], bk, s);
      }
      // scale by 1/sqrt(64), mask, stash to per-wave LDS in [m][k] layout.
#pragma unroll
      for (int r = 0; r < 8; ++r) {
        int m = r + 8 * half;
        int n = lane & 15;
        float v = s[r] * 0.125f;
        int mv = mask[((size_t)b * S_ + (q0 + m)) * S_ + kk + tile * 16 + n];
        if (mv == 0) v = -1e9f;
        sc[wave][m][tile * 16 + n] = v;
      }
    }

    // ---- reshape scores: C layout (LDS) -> bf16 A fragment (16x32) ----
    v16bf as;
#pragma unroll
    for (int e = 0; e < 16; ++e)
      as[e] = (__bf16)sc[wave][mr][a_kmap(e, half)];

    // ---- qkv += scores @ V band; B[k][d]: n = lane&15 (d), e -> k via a_kmap ----
#pragma unroll
    for (int t = 0; t < 4; ++t) {
      v16bf bv;
#pragma unroll
      for (int e = 0; e < 16; ++e)
        bv[e] = lv[buf][a_kmap(e, half)][t * 16 + mr];
      acc[t] = wmma_bf16(as, bv, acc[t]);
    }
    __syncthreads();
  }

  // ---- softmax over DK=64 per query row, store bf16 into X ----
#pragma unroll
  for (int r = 0; r < 8; ++r) {
    float mx = fmaxf(fmaxf(acc[0][r], acc[1][r]), fmaxf(acc[2][r], acc[3][r]));
#pragma unroll
    for (int off = 1; off < 16; off <<= 1)
      mx = fmaxf(mx, __shfl_xor(mx, off, 32));
    float e0 = __expf(acc[0][r] - mx);
    float e1 = __expf(acc[1][r] - mx);
    float e2 = __expf(acc[2][r] - mx);
    float e3 = __expf(acc[3][r] - mx);
    float sm = e0 + e1 + e2 + e3;
#pragma unroll
    for (int off = 1; off < 16; off <<= 1)
      sm += __shfl_xor(sm, off, 32);
    float inv = 1.0f / sm;

    int m = q0 + r + 8 * half;
    int n = lane & 15;
    size_t row = (size_t)(b * S_ + m) * D_ + (size_t)h * DK_;
    X[row +  0 + n] = (__bf16)(e0 * inv);
    X[row + 16 + n] = (__bf16)(e1 * inv);
    X[row + 32 + n] = (__bf16)(e2 * inv);
    X[row + 48 + n] = (__bf16)(e3 * inv);
  }
}

// ---------------------------------------------------------------------------
// Kernel 3: out = X @ Wo^T + bo (bf16 A, LDS-staged Wo, f32 out)
// ---------------------------------------------------------------------------
__global__ __launch_bounds__(256)
void outproj_kernel(const __bf16* __restrict__ X, const float* __restrict__ W,
                    const float* __restrict__ bias, float* __restrict__ out) {
  __shared__ float lw[2][64][32];

  const int tid  = threadIdx.x;
  const int lane = tid & 31;
  const int wave = tid >> 5;
  const int ng   = blockIdx.x & 15;
  const int mblk = blockIdx.x >> 4;
  const int m0   = (mblk * 8 + wave) * 16;
  const int n0   = ng * 64;
  const int half = lane >> 4, mr = lane & 15;

  auto stageW = [&](int kk, int buf) {
#pragma unroll
    for (int j = 0; j < 2; ++j) {
      int cid = tid + 256 * j;
      int row = cid >> 3, c = cid & 7;
      async_copy_b128(&lw[buf][row][c * 4],
                      &W[(size_t)(n0 + row) * D_ + kk + c * 4]);
    }
  };

  v8f acc[4] = {};
  stageW(0, 0);

  for (int kk = 0; kk < D_; kk += 32) {
    const int buf = (kk >> 5) & 1;
    if (kk + 32 < D_) { stageW(kk + 32, buf ^ 1); wait_async<2>(); }
    else              { wait_async<0>(); }
    __syncthreads();

    const __bf16* rowp = &X[(size_t)(m0 + mr) * D_ + kk];
    v8bf c0 = *(const v8bf*)(rowp + 8 * half);
    v8bf c1 = *(const v8bf*)(rowp + 16 + 8 * half);
    v16bf a;
#pragma unroll
    for (int e = 0; e < 8; ++e) { a[e] = c0[e]; a[e + 8] = c1[e]; }

#pragma unroll
    for (int t = 0; t < 4; ++t) {
      v16bf bfr;
#pragma unroll
      for (int e = 0; e < 16; ++e)
        bfr[e] = (__bf16)lw[buf][t * 16 + mr][a_kmap(e, half)];
      acc[t] = wmma_bf16(a, bfr, acc[t]);
    }
    __syncthreads();
  }

#pragma unroll
  for (int t = 0; t < 4; ++t) {
    int n = n0 + t * 16 + (lane & 15);
    float bn = bias[n];
#pragma unroll
    for (int r = 0; r < 8; ++r) {
      int m = m0 + r + 8 * half;
      out[(size_t)m * D_ + n] = acc[t][r] + bn;
    }
  }
}

// ---------------------------------------------------------------------------
// Launch. Inputs (setup_inputs dict order):
//  0:k 1:q 2:v 3:mask 4:Wq 5:bq 6:Wk 7:bk 8:Wv 9:bv 10:Wo 11:bo
// Workspace: qryb | keyb | valb | xb, each B*S*D bf16 (33.5 MB total)
// ---------------------------------------------------------------------------
extern "C" void kernel_launch(void* const* d_in, const int* in_sizes, int n_in,
                              void* d_out, int out_size, void* d_ws, size_t ws_size,
                              hipStream_t stream) {
  (void)in_sizes; (void)n_in; (void)out_size; (void)ws_size;
  const float* k    = (const float*)d_in[0];
  const float* q    = (const float*)d_in[1];
  const float* v    = (const float*)d_in[2];
  const int*   mask = (const int*)d_in[3];
  const float* Wq   = (const float*)d_in[4];
  const float* bq   = (const float*)d_in[5];
  const float* Wk   = (const float*)d_in[6];
  const float* bk   = (const float*)d_in[7];
  const float* Wv   = (const float*)d_in[8];
  const float* bv   = (const float*)d_in[9];
  const float* Wo   = (const float*)d_in[10];
  const float* bo   = (const float*)d_in[11];
  float* out = (float*)d_out;

  const size_t NE = (size_t)B_ * S_ * D_;  // 4,194,304 elements
  __bf16* qryb = (__bf16*)d_ws;
  __bf16* keyb = qryb + NE;
  __bf16* valb = keyb + NE;
  __bf16* xb   = valb + NE;

  proj_bf16_kernel<<<512, 256, 0, stream>>>(q, Wq, bq, qryb);
  proj_bf16_kernel<<<512, 256, 0, stream>>>(k, Wk, bk, keyb);
  proj_bf16_kernel<<<512, 256, 0, stream>>>(v, Wv, bv, valb);

  attn_fused_kernel<<<1024, 128, 0, stream>>>(qryb, keyb, valb, mask, xb);

  outproj_kernel<<<512, 256, 0, stream>>>(xb, Wo, bo, out);
}
